// FactorizedVQTokenizeWrapper_15152644620681
// MI455X (gfx1250) — matrix operand order repaired and checked by reading, not compiled, not run on hardware
//
#include <hip/hip_runtime.h>
#include <math.h>

#define B_    8
#define DIN   1024
#define T_    2048
#define DCB   64
#define V_    8192
#define NROWS (B_ * T_)   // 16384
#define VTILES (V_ / 16)  // 512

typedef float  v2f   __attribute__((ext_vector_type(2)));
typedef float  v8f   __attribute__((ext_vector_type(8)));
typedef __bf16 v16bf __attribute__((ext_vector_type(16)));
typedef __bf16 v8bf  __attribute__((ext_vector_type(8)));
typedef __bf16 v2bf  __attribute__((ext_vector_type(2)));
typedef int    v4i_t __attribute__((ext_vector_type(4)));

// ---------------------------------------------------------------------------
// CDNA5 async global->LDS staging (ASYNCcnt-tracked), with portable fallbacks.
// Probe-derived prototype: (int4* gaddr, int4* ldsaddr, imm offset, imm cpol).
// ---------------------------------------------------------------------------
#if __has_builtin(__builtin_amdgcn_global_load_async_to_lds_b128)
#define HAS_ASYNC_LDS 1
#else
#define HAS_ASYNC_LDS 0
#endif

#if HAS_ASYNC_LDS
typedef __attribute__((address_space(3))) v4i_t* lds_v4i_p;
#if __has_builtin(__builtin_amdgcn_s_wait_asynccnt)
#define WAIT_ASYNC0() __builtin_amdgcn_s_wait_asynccnt(0)
#else
#define WAIT_ASYNC0() asm volatile("s_wait_asynccnt 0" ::: "memory")
#endif
#else
#define WAIT_ASYNC0() ((void)0)
#endif

__device__ __forceinline__ void stage16(const void* g, void* l) {
#if HAS_ASYNC_LDS
  __builtin_amdgcn_global_load_async_to_lds_b128((v4i_t*)g, (lds_v4i_p)l, 0, 0);
#else
  *(uint4*)l = *(const uint4*)g;   // synchronous VGPR round-trip fallback
#endif
}

// ---------------------------------------------------------------------------
// Kernel 1: L2-normalize codebook rows, split each element into bf16 hi + lo.
// One wave32 per row: lane L owns elements 2L, 2L+1 (float2, fully coalesced).
// ---------------------------------------------------------------------------
__global__ __launch_bounds__(256) void k_norm_codebook(
    const float* __restrict__ cb,
    __bf16* __restrict__ cn_hi, __bf16* __restrict__ cn_lo) {
  const int lane = threadIdx.x & 31;
  const int wv   = threadIdx.x >> 5;
  const int v    = blockIdx.x * 8 + wv;

  const float2 c2 = ((const float2*)(cb + (size_t)v * DCB))[lane];
  float ss = c2.x * c2.x + c2.y * c2.y;
  #pragma unroll
  for (int m = 16; m >= 1; m >>= 1) ss += __shfl_xor(ss, m, 32);
  const float inv = 1.0f / fmaxf(sqrtf(ss), 1e-12f);

  const float x = c2.x * inv, y = c2.y * inv;
  const __bf16 xh = (__bf16)x, yh = (__bf16)y;
  const __bf16 xl = (__bf16)(x - (float)xh), yl = (__bf16)(y - (float)yh);
  v2bf hi, lo;
  hi[0] = xh; hi[1] = yh;
  lo[0] = xl; lo[1] = yl;
  ((v2bf*)(cn_hi + (size_t)v * DCB))[lane] = hi;
  ((v2bf*)(cn_lo + (size_t)v * DCB))[lane] = lo;
}

// ---------------------------------------------------------------------------
// Kernel 2: projection e[n,d] = sum_c z[b,c,t]*W[d,c] + bias[d], n = b*T + t.
// Tile: M=16 rows (n), N=64 (one 16-wide d-tile per wave, 4 waves), K=1024.
// Native fp32 WMMA 16x16x4 (exact fp32 math); output stored bf16 hi/lo split.
// ---------------------------------------------------------------------------
#if __has_builtin(__builtin_amdgcn_wmma_f32_16x16x4_f32)
__global__ __launch_bounds__(128) void k_project(
    const float* __restrict__ z, const float* __restrict__ W,
    const float* __restrict__ bias,
    __bf16* __restrict__ e_hi, __bf16* __restrict__ e_lo) {
  const int lane = threadIdx.x & 31;
  const int wv   = threadIdx.x >> 5;       // d-tile 0..3
  const int n0   = blockIdx.x * 16;        // T divisible by 16 -> single b
  const int b    = n0 / T_;
  const int t0   = n0 - b * T_;
  const int l16  = lane & 15;
  const int hb   = lane >> 4;              // lane half (K phase)
  const int d    = wv * 16 + l16;          // column owned by this lane slot

  // A frag: lane(l16,hb) reads z[b, k + 2*hb + {0,1}, t0 + l16]
  const float* aptr = z + (size_t)b * DIN * T_ + t0 + l16 + (size_t)(2 * hb) * T_;
  // B frag: lane reads W[d, k + 2*hb + {0,1}] (contiguous pair)
  const float* bptr = W + (size_t)d * DIN + 2 * hb;

  v8f acc = {};
  #pragma unroll 4
  for (int k = 0; k < DIN; k += 4) {
    v2f a;
    a.x = aptr[(size_t)k * T_];
    a.y = aptr[(size_t)k * T_ + T_];
    const v2f bb = *(const v2f*)(bptr + k);
    acc = __builtin_amdgcn_wmma_f32_16x16x4_f32(
        false, a, false, bb, (short)0, acc, false, false);
  }

  const float bv = bias[d];
  // C layout: VGPR r -> row m = r + 8*hb ; N = l16 -> column d
  #pragma unroll
  for (int r = 0; r < 8; ++r) {
    const float val = acc[r] + bv;
    const __bf16 h = (__bf16)val;
    const __bf16 l = (__bf16)(val - (float)h);
    const size_t idx = (size_t)(n0 + r + 8 * hb) * DCB + d;
    e_hi[idx] = h;
    e_lo[idx] = l;
  }
}
#else
__global__ __launch_bounds__(64) void k_project(
    const float* __restrict__ z, const float* __restrict__ W,
    const float* __restrict__ bias,
    __bf16* __restrict__ e_hi, __bf16* __restrict__ e_lo) {
  const int n = blockIdx.x;
  const int d = threadIdx.x;
  const int b = n / T_, t = n - b * T_;
  const float* zp = z + (size_t)b * DIN * T_ + t;
  const float* wp = W + (size_t)d * DIN;
  float acc = bias[d];
  for (int c = 0; c < DIN; ++c) acc = fmaf(zp[(size_t)c * T_], wp[c], acc);
  const __bf16 h = (__bf16)acc;
  const __bf16 l = (__bf16)(acc - (float)h);
  e_hi[(size_t)n * DCB + d] = h;
  e_lo[(size_t)n * DCB + d] = l;
}
#endif

// ---------------------------------------------------------------------------
// Kernel 3: scores = e @ cn^T (K=64) via split-bf16 WMMA, fused row argmax.
// Block = 64 rows: wave w owns rows n0+16w..+15 (A frags held in registers).
// All 4 waves share one cn fragment stream: groups of 4 v-tiles are staged
// into LDS with global_load_async_to_lds_b128 (double buffered, per-lane
// 48B slots), cutting cn L2 traffic 4x. 6 chained v_wmma_f32_16x16x32_bf16
// per tile (a_lo*b_hi + a_hi*b_lo + a_hi*b_hi). Running per-lane (max,idx),
// then an in-wave shfl_xor butterfly with first-index tie-break.
// ---------------------------------------------------------------------------
#define GTILES      4                         // v-tiles per staged group
#define NGROUPS     (VTILES / GTILES)         // 128
#define FRAG_STRIDE 48                        // bytes per lane slot (pad 32->48)
#define SLOT_BYTES  (32 * FRAG_STRIDE)        // 1536 B per (tile,matrix,chunk)
#define SLOTS_PG    (GTILES * 4)              // 16 slots per group
#define BUF_BYTES   (SLOTS_PG * SLOT_BYTES)   // 24576 B per buffer

__global__ __launch_bounds__(128) void k_argmax(
    const __bf16* __restrict__ e_hi, const __bf16* __restrict__ e_lo,
    const __bf16* __restrict__ cn_hi, const __bf16* __restrict__ cn_lo,
    int* __restrict__ out) {
  __shared__ __align__(16) char smem[2 * BUF_BYTES];   // 48 KB

  const int lane = threadIdx.x & 31;
  const int wv   = threadIdx.x >> 5;   // 0..3
  const int n0   = blockIdx.x * 64;
  const int l16  = lane & 15;
  const int hb   = lane >> 4;

  // --- preload this wave's A fragments: rows n0+16*wv .. +15, K=64
  //     16x32 bf16 A layout per chunk:
  //       lane<16 : elems[0..7]=K 0..7,  [8..15]=K 16..23
  //       lane>=16: elems[0..7]=K 8..15, [8..15]=K 24..31
  union VB { v16bf v; v8bf h[2]; };
  VB a_hi[2], a_lo[2];
  const size_t arow = (size_t)(n0 + wv * 16 + l16) * DCB;
  #pragma unroll
  for (int ch = 0; ch < 2; ++ch) {
    const int base = ch * 32 + hb * 8;
    a_hi[ch].h[0] = *(const v8bf*)(e_hi + arow + base);
    a_hi[ch].h[1] = *(const v8bf*)(e_hi + arow + base + 16);
    a_lo[ch].h[0] = *(const v8bf*)(e_lo + arow + base);
    a_lo[ch].h[1] = *(const v8bf*)(e_lo + arow + base + 16);
  }

  // --- stage helper: wave wv stages tile t=wv of group g into buffer parity p
  const __bf16* const srcs[2] = { cn_hi, cn_lo };
  auto stage_group = [&](int g) {
    const int p  = g & 1;
    const int v0 = (g * GTILES + wv) * 16;       // this wave's tile
    #pragma unroll
    for (int mat = 0; mat < 2; ++mat)
      #pragma unroll
      for (int ch = 0; ch < 2; ++ch)
        #pragma unroll
        for (int h = 0; h < 2; ++h) {
          const __bf16* gp = srcs[mat] +
              (size_t)(v0 + l16) * DCB + ch * 32 + hb * 8 + h * 16;
          char* lp = smem + p * BUF_BYTES +
              (wv * 4 + mat * 2 + ch) * SLOT_BYTES + lane * FRAG_STRIDE + h * 16;
          stage16(gp, lp);
        }
  };

  float bestS[8];
  int   bestV[8];
  #pragma unroll
  for (int r = 0; r < 8; ++r) { bestS[r] = -3.0e38f; bestV[r] = 0; }

  stage_group(0);

  for (int g = 0; g < NGROUPS; ++g) {
    WAIT_ASYNC0();          // my async stores for group g have landed in LDS
    __syncthreads();        // everyone's stores landed; g-1 consumers done
    if (g + 1 < NGROUPS) stage_group(g + 1);   // overlap DMA with WMMAs

    const size_t pbase = (size_t)(g & 1) * BUF_BYTES + (size_t)lane * FRAG_STRIDE;
    #pragma unroll
    for (int t = 0; t < GTILES; ++t) {
      v8f acc = {};
      #pragma unroll
      for (int ch = 0; ch < 2; ++ch) {
        VB bh, bl;
        const size_t oh = pbase + (size_t)(t * 4 + ch) * SLOT_BYTES;       // mat 0
        const size_t ol = oh + 2 * SLOT_BYTES;                             // mat 1
        bh.h[0] = *(const v8bf*)(smem + oh);
        bh.h[1] = *(const v8bf*)(smem + oh + 16);
        bl.h[0] = *(const v8bf*)(smem + ol);
        bl.h[1] = *(const v8bf*)(smem + ol + 16);
        acc = __builtin_amdgcn_wmma_f32_16x16x32_bf16(
            false, a_lo[ch].v, false, bh.v, (short)0, acc, false, false);
        acc = __builtin_amdgcn_wmma_f32_16x16x32_bf16(
            false, a_hi[ch].v, false, bl.v, (short)0, acc, false, false);
        acc = __builtin_amdgcn_wmma_f32_16x16x32_bf16(
            false, a_hi[ch].v, false, bh.v, (short)0, acc, false, false);
      }
      const int vcol = (g * GTILES + t) * 16 + l16;   // C layout: N = l16
      #pragma unroll
      for (int r = 0; r < 8; ++r) {
        const float s = acc[r];
        if (s > bestS[r]) { bestS[r] = s; bestV[r] = vcol; }  // strict > -> lowest v
      }
    }
  }

  // --- in-wave argmax reduction across the 16 column slots of each row.
  //     Rows: lanes 0-15 hold m=r, lanes 16-31 hold m=r+8 (C layout).
  #pragma unroll
  for (int r = 0; r < 8; ++r) {
    float s = bestS[r];
    int   v = bestV[r];
    #pragma unroll
    for (int m = 8; m >= 1; m >>= 1) {
      const float so = __shfl_xor(s, m, 32);
      const int   vo = __shfl_xor(v, m, 32);
      if (so > s || (so == s && vo < v)) { s = so; v = vo; }
    }
    if (l16 == 0) out[n0 + wv * 16 + r + 8 * hb] = v;
  }
}

// ---------------------------------------------------------------------------
extern "C" void kernel_launch(void* const* d_in, const int* in_sizes, int n_in,
                              void* d_out, int out_size, void* d_ws, size_t ws_size,
                              hipStream_t stream) {
  const float* z    = (const float*)d_in[0];   // [8, 1024, 2048]
  const float* W    = (const float*)d_in[1];   // [64, 1024]
  const float* bias = (const float*)d_in[2];   // [64]
  const float* cb   = (const float*)d_in[3];   // [8192, 64]

  char* ws = (char*)d_ws;
  __bf16* cn_hi = (__bf16*)(ws);                          // 1 MB
  __bf16* cn_lo = (__bf16*)(ws + ((size_t)1 << 20));      // 1 MB
  __bf16* e_hi  = (__bf16*)(ws + ((size_t)2 << 20));      // 2 MB
  __bf16* e_lo  = (__bf16*)(ws + ((size_t)4 << 20));      // 2 MB

  k_norm_codebook<<<V_ / 8, 256, 0, stream>>>(cb, cn_hi, cn_lo);

#if __has_builtin(__builtin_amdgcn_wmma_f32_16x16x4_f32)
  k_project<<<NROWS / 16, 128, 0, stream>>>(z, W, bias, e_hi, e_lo);
#else
  k_project<<<NROWS, 64, 0, stream>>>(z, W, bias, e_hi, e_lo);
#endif

  k_argmax<<<NROWS / 64, 128, 0, stream>>>(e_hi, e_lo, cn_hi, cn_lo, (int*)d_out);
}
